// PointcloudImageFusion_20933670601444
// MI455X (gfx1250) — matrix operand order repaired and checked by reading, not compiled, not run on hardware
//
#include <hip/hip_runtime.h>
#include <hip/hip_bf16.h>
#include <stdint.h>

// ---------------- types for CDNA5 WMMA ----------------
typedef __bf16 bf16_t;
typedef __attribute__((ext_vector_type(16))) __bf16 v16bf;
typedef __attribute__((ext_vector_type(8)))  float  v8f;

static constexpr int kB   = 32;
static constexpr int kN   = 256;
static constexpr int kIMG = 1024;
static constexpr int kPC  = 768;
static constexpr int kVIS = 512;
static constexpr int kTOK = kB * kN;         // 8192 tokens
static constexpr float kTHRESH = 0.3f;
static constexpr float kLNEPS  = 1e-5f;

// ---------------- WMMA fragment loaders ----------------
// A (16x32 bf16, MxK): lane L row M=L%16. lane<16: K = {0..7, 16..23}; lane>=16: K = {8..15, 24..31}.
// caller passes p = rowBase + kbase + ((lane&16)?8:0)
static __device__ inline v16bf loadA_bf(const bf16_t* p) {
  union { uint4 u[2]; v16bf v; } U;
  U.u[0] = *(const uint4*)(p);        // 8 bf16, first K-run
  U.u[1] = *(const uint4*)(p + 16);   // 8 bf16, second K-run (+16 in K)
  return U.v;
}
// B (32x16 bf16, KxN): lane L col N=L%16, K = (L/16)*16 + e, e=0..15 contiguous along K of W-row.
// caller passes p = W + (tileN + lane%16)*K + kbase + ((lane&16)?16:0)
static __device__ inline v16bf loadB_bf(const bf16_t* p) {
  union { uint4 u[2]; v16bf v; } U;
  U.u[0] = *(const uint4*)(p);
  U.u[1] = *(const uint4*)(p + 8);
  return U.v;
}
// Same fragments built from fp32 source with in-register bf16 conversion.
static __device__ inline v16bf loadA_f32(const float* p) {
  v16bf a;
#pragma unroll
  for (int e = 0; e < 8; ++e) a[e] = (bf16_t)p[e];
#pragma unroll
  for (int e = 0; e < 8; ++e) a[8 + e] = (bf16_t)p[8 + e];
  return a;
}
static __device__ inline v16bf loadB_f32(const float* p) {
  v16bf v;
#pragma unroll
  for (int e = 0; e < 16; ++e) v[e] = (bf16_t)p[e];
  return v;
}

#define WMMA_BF16(a, b, c) \
  __builtin_amdgcn_wmma_f32_16x16x32_bf16(false, (a), false, (b), (short)0, (c), false, false)

// ---------------- 1) fp32 -> bf16 weight conversion ----------------
__global__ void f32_to_bf16_kernel(const float* __restrict__ src, bf16_t* __restrict__ dst, int n) {
  int i = blockIdx.x * blockDim.x + threadIdx.x;
  int stride = gridDim.x * blockDim.x;
  for (; i < n; i += stride) dst[i] = (bf16_t)src[i];
}

// ---------------- 2) reciprocal row norms over concat(A,B2) ----------------
// one wave per token; 8 waves per block
__global__ void rnorm_kernel(const float* __restrict__ A, int Da,
                             const float* __restrict__ B2, int Db,
                             float* __restrict__ out) {
  int wave = threadIdx.x >> 5;
  int lane = threadIdx.x & 31;
  int t = blockIdx.x * 8 + wave;
  if (t >= kTOK) return;
  const float* pa = A + (size_t)t * Da;
  const float* pb = B2 + (size_t)t * Db;
  float s = 0.f;
  for (int c = lane; c < Da; c += 32) { float v = pa[c]; s += v * v; }
  for (int c = lane; c < Db; c += 32) { float v = pb[c]; s += v * v; }
#pragma unroll
  for (int o = 16; o > 0; o >>= 1) s += __shfl_xor(s, o, 32);
  if (lane == 0) out[t] = 1.f / fmaxf(sqrtf(s), 1e-8f);
}

// ---------------- 3) cosine-sim GEMM via WMMA bf16 ----------------
// sim[b,i,j] = rnv[b,i]*rnp[b,j] * ( vis[b,i,:] . pvis[b,j,:] ), K = 1024 (img) + 768 (pc)
// block = 128 (4 waves), wave w handles tile row ti = blockIdx.y*64 + w*16
__global__ void sim_wmma_kernel(const float* __restrict__ imgF,  const float* __restrict__ pcF,
                                const float* __restrict__ pimgF, const float* __restrict__ ppcF,
                                const float* __restrict__ rnv,   const float* __restrict__ rnp,
                                float* __restrict__ sim) {
  int lane = threadIdx.x & 31;
  int wave = threadIdx.x >> 5;
  int tj = blockIdx.x * 16;
  int ti = blockIdx.y * 64 + wave * 16;
  int b  = blockIdx.z;

  int ar = ti + (lane & 15);          // A row (query token within batch)
  int bc = tj + (lane & 15);          // B col (prev token within batch)

  const float* a0 = imgF  + ((size_t)b * kN + ar) * kIMG + ((lane & 16) ? 16 : 0);
  const float* b0 = pimgF + ((size_t)b * kN + bc) * kIMG + ((lane & 16) ? 16 : 0);
  const float* a1 = pcF   + ((size_t)b * kN + ar) * kPC  + ((lane & 16) ? 16 : 0);
  const float* b1 = ppcF  + ((size_t)b * kN + bc) * kPC  + ((lane & 16) ? 16 : 0);

  v8f acc = {};
  for (int kb = 0; kb < kIMG; kb += 32) {
    v16bf af = loadA_f32(a0 + kb);
    v16bf bf = loadB_f32(b0 + kb);
    acc = WMMA_BF16(af, bf, acc);
  }
  for (int kb = 0; kb < kPC; kb += 32) {
    v16bf af = loadA_f32(a1 + kb);
    v16bf bf = loadB_f32(b1 + kb);
    acc = WMMA_BF16(af, bf, acc);
  }

  float sj = rnp[b * kN + bc];
#pragma unroll
  for (int r = 0; r < 8; ++r) {
    int row = ti + r + ((lane & 16) ? 8 : 0);
    sim[((size_t)b * kN + row) * kN + bc] = acc[r] * rnv[b * kN + row] * sj;
  }
}

// ---------------- 4) greedy sequential matching ----------------
// one block per batch; 256 threads; sequential over rows i (matches lax.scan semantics)
__global__ void match_kernel(const float* __restrict__ sim,
                             const unsigned char* __restrict__ mask,   // numpy bool, 1 byte
                             int* __restrict__ idx) {
  int b = blockIdx.x;
  int j = threadIdx.x;
  __shared__ float sv[256];
  __shared__ int   sj[256];
  __shared__ unsigned char visited[256];
  visited[j] = 0;
  __syncthreads();
  const float* simb = sim + (size_t)b * kN * kN;
  const unsigned char* mb = mask + (size_t)b * kN * kN;
  for (int i = 0; i < kN; ++i) {
    float s = -__builtin_inff();
    if (mb[i * kN + j] && !visited[j]) s = simb[i * kN + j];
    sv[j] = s; sj[j] = j;
    __syncthreads();
    for (int off = 128; off > 0; off >>= 1) {
      if (j < off) {
        float a = sv[j], c = sv[j + off];
        if (c > a || (c == a && sj[j + off] < sj[j])) { sv[j] = c; sj[j] = sj[j + off]; }
      }
      __syncthreads();
    }
    if (j == 0) {
      float best = sv[0]; int bj = sj[0];
      bool ok = (best >= kTHRESH) && (best > -1.0f);
      idx[b * kN + i] = ok ? bj : -1;
      if (ok) visited[bj] = 1;
    }
    __syncthreads();
  }
}

// ---------------- 5) gather + concat -> bf16 GEMM input ----------------
__global__ void gather_concat_kernel(const float* __restrict__ cur,
                                     const float* __restrict__ prev,
                                     const int* __restrict__ idx,
                                     bf16_t* __restrict__ out, int D) {
  int t = blockIdx.x;
  int id = idx[t];
  int prow = ((t >> 8) << 8) + (id < 0 ? 0 : id);
  float m = (id >= 0) ? 1.f : 0.f;
  const float* pc_ = cur  + (size_t)t    * D;
  const float* pp  = prev + (size_t)prow * D;
  bf16_t* po = out + (size_t)t * (2 * D);
  for (int c = threadIdx.x; c < D; c += blockDim.x) {
    po[c]     = (bf16_t)pc_[c];
    po[D + c] = (bf16_t)(pp[c] * m);
  }
}

__global__ void gather_spatial_kernel(const float* __restrict__ sp,
                                      const int* __restrict__ idx,
                                      float* __restrict__ out) {
  int t = blockIdx.x * blockDim.x + threadIdx.x;
  if (t >= kTOK) return;
  int id = idx[t];
  int prow = ((t >> 8) << 8) + (id < 0 ? 0 : id);
  float m = (id >= 0) ? 1.f : 0.f;
  out[t * 3 + 0] = sp[prow * 3 + 0] * m;
  out[t * 3 + 1] = sp[prow * 3 + 1] * m;
  out[t * 3 + 2] = sp[prow * 3 + 2] * m;
}

// ---------------- 6) WMMA GEMM with async-LDS A staging ----------------
// y = act(x @ W^T + bias). X bf16 [8192, K], W bf16 [Nout, K] row-major.
// block = 128 threads (4 waves) owns a 64-row x 128-col output macro-tile.
// The 64x32 A tile for the current K-step is staged once per block into LDS with
// GLOBAL_LOAD_ASYNC_TO_LDS_B128 (ASYNCcnt) and consumed by all 4 waves via ds_load.
// Each wave: 4 M-tiles x 2 N-tiles = 8 WMMAs per K-step.
// grid = (Nout/128, 8192/64)
__global__ void gemm_bias_act(const bf16_t* __restrict__ X, const bf16_t* __restrict__ W,
                              const float* __restrict__ bias, void* __restrict__ outp,
                              int K, int ldOut, int relu, int outBf16) {
  __shared__ __align__(16) bf16_t tileA[64 * 32];   // 4 KB
  int tid  = threadIdx.x;
  int lane = tid & 31;
  int wave = tid >> 5;
  int colBase = blockIdx.x * 128 + wave * 32;       // wave covers 2 N tiles
  int rowBase = blockIdx.y * 64;

  const bf16_t* w0 = W + (size_t)(colBase + (lane & 15)) * K + ((lane & 16) ? 16 : 0);
  const bf16_t* w1 = w0 + (size_t)16 * K;

  // staging: 4KB tile = 256 x 16B chunks; each of 128 threads copies chunks 2*tid, 2*tid+1.
  // chunk c -> tile row c/4, K offset (c%4)*8 elements.
  int c0 = tid * 2;
  const bf16_t* g0 = X + (size_t)(rowBase + (c0 >> 2)) * K + (c0 & 3) * 8;
  const bf16_t* g1 = X + (size_t)(rowBase + ((c0 + 1) >> 2)) * K + ((c0 + 1) & 3) * 8;
  // LDS byte offsets (flat-pointer low 32 bits == LDS aperture offset, ISA 10.2)
  unsigned l0 = (unsigned)(size_t)(&tileA[c0 * 8]);
  unsigned l1 = (unsigned)(size_t)(&tileA[c0 * 8 + 8]);

  // per-lane LDS read base for A fragments
  const bf16_t* lp = tileA + (size_t)(lane & 15) * 32 + ((lane & 16) ? 16 : 0);

  v8f acc[8] = {};   // [m] for n-tile0, [4+m] for n-tile1
  for (int kb = 0; kb < K; kb += 32) {
    __syncthreads();   // previous iteration's LDS reads complete before overwrite
    {
      uint64_t a0 = (uint64_t)(size_t)(g0 + kb);
      uint64_t a1 = (uint64_t)(size_t)(g1 + kb);
      asm volatile("global_load_async_to_lds_b128 %0, %1, off"
                   :: "v"(l0), "v"(a0) : "memory");
      asm volatile("global_load_async_to_lds_b128 %0, %1, off"
                   :: "v"(l1), "v"(a1) : "memory");
      asm volatile("s_wait_asynccnt 0" ::: "memory");
    }
    __syncthreads();   // all waves' async copies now visible in LDS

    __builtin_prefetch((const void*)(w0 + kb + 256), 0, 0);   // global_prefetch weight stream
    v16bf b0 = loadB_bf(w0 + kb);
    v16bf b1 = loadB_bf(w1 + kb);
#pragma unroll
    for (int m = 0; m < 4; ++m) {
      // ds_load_b128 x2: two 8-element K-runs of the A fragment
      union { uint4 u[2]; v16bf v; } U;
      const bf16_t* pa = lp + (size_t)m * 16 * 32;
      U.u[0] = *(const uint4*)(pa);
      U.u[1] = *(const uint4*)(pa + 8);
      acc[m]     = WMMA_BF16(U.v, b0, acc[m]);
      acc[4 + m] = WMMA_BF16(U.v, b1, acc[4 + m]);
    }
  }

  int col0 = colBase + (lane & 15);
  float bv0 = bias[col0], bv1 = bias[col0 + 16];
#pragma unroll
  for (int m = 0; m < 4; ++m) {
#pragma unroll
    for (int r = 0; r < 8; ++r) {
      int row = rowBase + m * 16 + r + ((lane & 16) ? 8 : 0);
      float v0 = acc[m][r] + bv0;
      float v1 = acc[4 + m][r] + bv1;
      if (relu) { v0 = fmaxf(v0, 0.f); v1 = fmaxf(v1, 0.f); }
      if (outBf16) {
        bf16_t* o = (bf16_t*)outp;
        o[(size_t)row * ldOut + col0]      = (bf16_t)v0;
        o[(size_t)row * ldOut + col0 + 16] = (bf16_t)v1;
      } else {
        float* o = (float*)outp;
        o[(size_t)row * ldOut + col0]      = v0;
        o[(size_t)row * ldOut + col0 + 16] = v1;
      }
    }
  }
}

// ---------------- 7) LayerNorm over D=512, one block per row ----------------
__global__ void layernorm_kernel(const float* __restrict__ x,
                                 const float* __restrict__ g, const float* __restrict__ be,
                                 void* __restrict__ outp, int ldOut, int colOff, int outBf16) {
  __shared__ float red[256];
  int row = blockIdx.x, tid = threadIdx.x;
  float v0 = x[(size_t)row * kVIS + tid];
  float v1 = x[(size_t)row * kVIS + 256 + tid];
  red[tid] = v0 + v1;
  __syncthreads();
  for (int o = 128; o > 0; o >>= 1) { if (tid < o) red[tid] += red[tid + o]; __syncthreads(); }
  float mu = red[0] / (float)kVIS;
  __syncthreads();
  float d0 = v0 - mu, d1 = v1 - mu;
  red[tid] = d0 * d0 + d1 * d1;
  __syncthreads();
  for (int o = 128; o > 0; o >>= 1) { if (tid < o) red[tid] += red[tid + o]; __syncthreads(); }
  float rs = rsqrtf(red[0] / (float)kVIS + kLNEPS);
  float y0 = d0 * rs * g[tid]       + be[tid];
  float y1 = d1 * rs * g[256 + tid] + be[256 + tid];
  if (outBf16) {
    bf16_t* o_ = (bf16_t*)outp;
    o_[(size_t)row * ldOut + colOff + tid]       = (bf16_t)y0;
    o_[(size_t)row * ldOut + colOff + 256 + tid] = (bf16_t)y1;
  } else {
    float* o_ = (float*)outp;
    o_[(size_t)row * ldOut + colOff + tid]       = y0;
    o_[(size_t)row * ldOut + colOff + 256 + tid] = y1;
  }
}

// ---------------- host side ----------------
extern "C" void kernel_launch(void* const* d_in, const int* in_sizes, int n_in,
                              void* d_out, int out_size, void* d_ws, size_t ws_size,
                              hipStream_t stream) {
  // inputs (setup_inputs order; params leaves in jax tree order: fin, fin_ln, img, img_ln, pc, pc_ln)
  const float* imgF  = (const float*)d_in[0];
  const float* pcF   = (const float*)d_in[1];
  const float* pimgF = (const float*)d_in[2];
  const float* ppcF  = (const float*)d_in[3];
  const unsigned char* mask = (const unsigned char*)d_in[4];   // numpy bool (1 byte/elem)
  const float* pspat = (const float*)d_in[5];
  const float* finW0 = (const float*)d_in[6];   const float* finb0 = (const float*)d_in[7];
  const float* finW1 = (const float*)d_in[8];   const float* finb1 = (const float*)d_in[9];
  const float* finG  = (const float*)d_in[10];  const float* finBe = (const float*)d_in[11];
  const float* imgW0 = (const float*)d_in[12];  const float* imgb0 = (const float*)d_in[13];
  const float* imgW1 = (const float*)d_in[14];  const float* imgb1 = (const float*)d_in[15];
  const float* imgW2 = (const float*)d_in[16];  const float* imgb2 = (const float*)d_in[17];
  const float* imgG  = (const float*)d_in[18];  const float* imgBe = (const float*)d_in[19];
  const float* pcW0  = (const float*)d_in[20];  const float* pcb0  = (const float*)d_in[21];
  const float* pcW1  = (const float*)d_in[22];  const float* pcb1  = (const float*)d_in[23];
  const float* pcW2  = (const float*)d_in[24];  const float* pcb2  = (const float*)d_in[25];
  const float* pcG   = (const float*)d_in[26];  const float* pcBe  = (const float*)d_in[27];

  char* ws = (char*)d_ws;
  // workspace arena (with lifetime-based reuse)
  float*  sim   = (float*) (ws + 0);                         //  8,388,608 B
  int*    idx   = (int*)   (ws + 8388608);                   //     32,768 B
  float*  rnv   = (float*) (ws + 8421376);                   //     32,768 B
  float*  rnp   = (float*) (ws + 8454144);                   //     32,768 B
  bf16_t* wA    = (bf16_t*)(ws + 8486912);                   // 13,238,272 B  bf16 weights
  bf16_t* xImg  = (bf16_t*)(ws + 21725184);                  // 33,554,432 B  [8192,2048]
  bf16_t* hB    = (bf16_t*)(ws + 21725184);                  // reuse (xImg dead after img layer0)
  float*  preLN = (float*) (ws + 21725184 + 16777216);       // reuse (f32 [8192,512])
  bf16_t* xPc   = (bf16_t*)(ws + 55279616);                  // 25,165,824 B  [8192,1536]
  bf16_t* hA    = (bf16_t*)(ws + 80445440);                  // 16,777,216 B
  bf16_t* xFin  = (bf16_t*)(ws + 97222656);                  // 16,777,216 B  [8192,1024]

  // bf16 weight sub-offsets (elements)
  bf16_t* bImgW0 = wA + 0;
  bf16_t* bImgW1 = wA + 2097152;
  bf16_t* bImgW2 = wA + 3145728;
  bf16_t* bPcW0  = wA + 3670016;
  bf16_t* bPcW1  = wA + 4849664;
  bf16_t* bPcW2  = wA + 5439488;
  bf16_t* bFinW0 = wA + 5832704;
  bf16_t* bFinW1 = wA + 6356992;

  auto cvt = [&](const float* s, bf16_t* d, int n) {
    int grid = (n + 1023) / 1024; if (grid > 4096) grid = 4096;
    f32_to_bf16_kernel<<<grid, 256, 0, stream>>>(s, d, n);
  };
  cvt(imgW0, bImgW0, 1024 * 2048);
  cvt(imgW1, bImgW1, 1024 * 1024);
  cvt(imgW2, bImgW2, 512 * 1024);
  cvt(pcW0,  bPcW0,  768 * 1536);
  cvt(pcW1,  bPcW1,  768 * 768);
  cvt(pcW2,  bPcW2,  512 * 768);
  cvt(finW0, bFinW0, 512 * 1024);
  cvt(finW1, bFinW1, 512 * 512);

  // reciprocal norms of concat rows
  rnorm_kernel<<<kTOK / 8, 256, 0, stream>>>(imgF, kIMG, pcF, kPC, rnv);
  rnorm_kernel<<<kTOK / 8, 256, 0, stream>>>(pimgF, kIMG, ppcF, kPC, rnp);

  // cosine similarity via WMMA
  sim_wmma_kernel<<<dim3(kN / 16, kN / 64, kB), 128, 0, stream>>>(
      imgF, pcF, pimgF, ppcF, rnv, rnp, sim);

  // greedy matching
  match_kernel<<<kB, 256, 0, stream>>>(sim, mask, idx);

  // gather + concat (fp32 -> bf16) and new_spatial
  float* out_x  = (float*)d_out;                  // [8192,512]
  float* out_sp = out_x + (size_t)kTOK * kVIS;    // [8192,3]
  gather_concat_kernel<<<kTOK, 256, 0, stream>>>(imgF, pimgF, idx, xImg, kIMG);
  gather_concat_kernel<<<kTOK, 256, 0, stream>>>(pcF, ppcF, idx, xPc, kPC);
  gather_spatial_kernel<<<(kTOK + 255) / 256, 256, 0, stream>>>(pspat, idx, out_sp);

  // img branch: 2048 ->relu-> 1024 -> 1024 -> 512 -> LN -> xFin[:, :512]
  gemm_bias_act<<<dim3(1024 / 128, kTOK / 64), 128, 0, stream>>>(xImg, bImgW0, imgb0, hA,    2048, 1024, 1, 1);
  gemm_bias_act<<<dim3(1024 / 128, kTOK / 64), 128, 0, stream>>>(hA,   bImgW1, imgb1, hB,    1024, 1024, 0, 1);
  gemm_bias_act<<<dim3(512  / 128, kTOK / 64), 128, 0, stream>>>(hB,   bImgW2, imgb2, preLN, 1024, 512,  0, 0);
  layernorm_kernel<<<kTOK, 256, 0, stream>>>(preLN, imgG, imgBe, xFin, 1024, 0, 1);

  // pc branch: 1536 ->relu-> 768 -> 768 -> 512 -> LN -> xFin[:, 512:]
  gemm_bias_act<<<dim3(768 / 128, kTOK / 64), 128, 0, stream>>>(xPc, bPcW0, pcb0, hA,    1536, 768, 1, 1);
  gemm_bias_act<<<dim3(768 / 128, kTOK / 64), 128, 0, stream>>>(hA,  bPcW1, pcb1, hB,     768, 768, 0, 1);
  gemm_bias_act<<<dim3(512 / 128, kTOK / 64), 128, 0, stream>>>(hB,  bPcW2, pcb2, preLN,  768, 512, 0, 0);
  layernorm_kernel<<<kTOK, 256, 0, stream>>>(preLN, pcG, pcBe, xFin, 1024, 512, 1);

  // final: 1024 ->relu-> 512 -> 512 -> LN -> d_out
  gemm_bias_act<<<dim3(512 / 128, kTOK / 64), 128, 0, stream>>>(xFin, bFinW0, finb0, hA,    1024, 512, 1, 1);
  gemm_bias_act<<<dim3(512 / 128, kTOK / 64), 128, 0, stream>>>(hA,   bFinW1, finb1, preLN,  512, 512, 0, 0);
  layernorm_kernel<<<kTOK, 256, 0, stream>>>(preLN, finG, finBe, out_x, 512, 0, 0);
}
